// COTREC_70342974374116
// MI455X (gfx1250) — matrix operand order repaired and checked by reading, not compiled, not run on hardware
//
#include <hip/hip_runtime.h>
#include <hip/hip_bf16.h>
#include <math.h>

// MI455X / gfx1250: wave32, WMMA 16x16x32 f16->f32 accumulate.
// SpMM is the bandwidth hog (~2.9GB -> ~125us @ 23.3TB/s); all dense GEMMs
// run on v_wmma_f32_16x16x32_f16 with f16 operands staged for contiguous
// per-lane fragment loads.

typedef _Float16 v16h __attribute__((ext_vector_type(16)));
typedef float    v8f  __attribute__((ext_vector_type(8)));

#define N_NODE 100000
#define EMBD   112
#define BATCH  512
#define SEQL   50
#define NNZE   1600000
#define LPAD   64       // sessions padded 50 -> 64 rows
#define KPAD   128      // inner dim padded 112 -> 128

// ---------------- fragment loaders (16-bit operand VGPR layout) ----------------
// A (MxK row-major): lanes 0-15 row=lane, halves = K[k0+0..15];
// lanes 16-31 row=lane-16, halves = K[k0+16..31].
__device__ __forceinline__ v16h load_a_frag(const _Float16* __restrict__ A, int lda,
                                            int m0, int k0, int lane) {
    const _Float16* p = A + (long long)(m0 + (lane & 15)) * lda + k0 + ((lane & 16) ? 16 : 0);
    v16h r;
#pragma unroll
    for (int h = 0; h < 16; ++h) r[h] = p[h];
    return r;
}
// B given as BT (NxK row-major): lane column = n0+(lane&15); 16 contiguous K halves,
// lanes 0-15 -> K[k0..k0+15], lanes 16-31 -> K[k0+16..k0+31].
__device__ __forceinline__ v16h load_bt_frag(const _Float16* __restrict__ BT, int ldb,
                                             int n0, int k0, int lane) {
    const _Float16* p = BT + (long long)(n0 + (lane & 15)) * ldb + k0 + ((lane & 16) ? 16 : 0);
    v16h r;
#pragma unroll
    for (int h = 0; h < 16; ++h) r[h] = p[h];
    return r;
}

// ---------------- utility kernels ----------------
__global__ void k_zero16(float4* p, long long n) {
    long long i = (long long)blockIdx.x * blockDim.x + threadIdx.x;
    if (i < n) p[i] = make_float4(0.f, 0.f, 0.f, 0.f);
}

// scatter-add SpMM: one thread per (nnz, float4-chunk); 28 chunks of 4 floats = 112
__global__ void k_spmm(const float* __restrict__ emb_in, const float* __restrict__ vals,
                       const int* __restrict__ rows, const int* __restrict__ cols,
                       float* __restrict__ emb_out) {
    long long gid = (long long)blockIdx.x * blockDim.x + threadIdx.x;
    if (gid >= (long long)NNZE * 28) return;
    int e = (int)(gid / 28);
    int c = (int)(gid % 28);
    int col = cols[e], row = rows[e];
    float v = vals[e];
    const float4 s = *(const float4*)(emb_in + (long long)col * EMBD + c * 4);
    float* dst = emb_out + (long long)row * EMBD + c * 4;
    atomicAdd(dst + 0, v * s.x);
    atomicAdd(dst + 1, v * s.y);
    atomicAdd(dst + 2, v * s.z);
    atomicAdd(dst + 3, v * s.w);
}

__global__ void k_combine(const float* __restrict__ e0, const float* __restrict__ e1,
                          const float* __restrict__ e2, float* __restrict__ item) {
    long long i = (long long)blockIdx.x * blockDim.x + threadIdx.x;
    if (i < (long long)N_NODE * EMBD) item[i] = (e0[i] + e1[i] + e2[i]) * (1.f / 3.f);
}

// W_q/W_k -> BT (n,k) f16 padded; w_sess[i] used as B^T directly (s@w.T => BT=w)
__global__ void k_pack_weights(const float* __restrict__ Wq, const float* __restrict__ Wk,
                               const float* __restrict__ ws,
                               _Float16* __restrict__ WqT, _Float16* __restrict__ WkT,
                               _Float16* __restrict__ wT) {
    int i = blockIdx.x * blockDim.x + threadIdx.x;
    if (i >= EMBD * EMBD) return;
    int k = i / EMBD, n = i % EMBD;
    WqT[n * KPAD + k] = (_Float16)Wq[k * EMBD + n];
    WkT[n * KPAD + k] = (_Float16)Wk[k * EMBD + n];
    wT[n * KPAD + k]                = (_Float16)ws[n * EMBD + k];
    wT[EMBD * KPAD + n * KPAD + k]  = (_Float16)ws[EMBD * EMBD + n * EMBD + k];
}

// seq gather: [B][64(pad)][112] f32 + [B*64][128(pad)] f16
__global__ void k_gather_seq(const float* __restrict__ item, const int* __restrict__ sess_item,
                             float* __restrict__ seqf, _Float16* __restrict__ seq16) {
    int blk = blockIdx.x;              // 512*50 blocks
    int b = blk / SEQL, l = blk % SEQL;
    int d = threadIdx.x;
    if (d >= EMBD) return;
    int idx = sess_item[b * SEQL + l];
    float v = (idx > 0) ? item[(long long)(idx - 1) * EMBD + d] : 0.f;
    long long r = (long long)b * LPAD + l;
    seqf[r * EMBD + d] = v;
    seq16[r * KPAD + d] = (_Float16)v;
}

// generic (batched) WMMA GEMM: D = post(A @ BT^T); one 16x16 tile per wave.
// flags bit0: sigmoid; then multiply by scale. Optional f32 and/or f16 outputs.
__global__ void k_wmma_gemm(const _Float16* __restrict__ A, int lda, long long sA,
                            const _Float16* __restrict__ BT, int ldb, long long sB,
                            float* __restrict__ C, int ldc, long long sC,
                            _Float16* __restrict__ Ch, int ldch, long long sCh,
                            int M, int N, int K, int nbatch, int flags, float scale) {
    int wave = (int)((blockIdx.x * (long long)blockDim.x + threadIdx.x) >> 5);
    int lane = threadIdx.x & 31;
    int tilesN = N >> 4;
    int tilesPer = (M >> 4) * tilesN;
    if (wave >= tilesPer * nbatch) return;   // wave-uniform: EXEC stays all-ones
    int b = wave / tilesPer;
    int t = wave % tilesPer;
    int m0 = (t / tilesN) << 4;
    int n0 = (t % tilesN) << 4;
    const _Float16* Ab = A + (long long)b * sA;
    const _Float16* Bb = BT + (long long)b * sB;
    v8f acc = {0.f, 0.f, 0.f, 0.f, 0.f, 0.f, 0.f, 0.f};
    for (int k0 = 0; k0 < K; k0 += 32) {
        v16h af = load_a_frag(Ab, lda, m0, k0, lane);
        v16h bf = load_bt_frag(Bb, ldb, n0, k0, lane);
        acc = __builtin_amdgcn_wmma_f32_16x16x32_f16(false, af, false, bf,
                                                     (short)0, acc, false, false);
    }
    int col = n0 + (lane & 15);
    int rbase = m0 + ((lane & 16) ? 8 : 0);
#pragma unroll
    for (int v = 0; v < 8; ++v) {
        float x = acc[v];
        if (flags & 1) x = 1.f / (1.f + __expf(-x));
        x *= scale;
        int row = rbase + v;
        if (C)  C [(long long)b * sC  + (long long)row * ldc  + col] = x;
        if (Ch) Ch[(long long)b * sCh + (long long)row * ldch + col] = (_Float16)x;
    }
}

// per-batch: mask C, alpha = rowsum/len, masked softmax, pool seq -> seqh, acc2
__global__ void k_attn_post(const float* __restrict__ Cbuf, const float* __restrict__ seqf,
                            const float* __restrict__ mask, const int* __restrict__ slen,
                            float* __restrict__ seqh, float* __restrict__ acc2) {
    int b = blockIdx.x, tid = threadIdx.x;
    __shared__ float alpha[LPAD];
    __shared__ float beta[LPAD];
    __shared__ float inv_sum;
    const float* Cb = Cbuf + (long long)b * LPAD * LPAD;
    float len = (float)slen[b];
    if (tid < SEQL) {
        float ml = mask[b * SEQL + tid];
        float s = 0.f;
        for (int m = 0; m < SEQL; ++m) {
            if (m == tid) continue;
            s += Cb[tid * LPAD + m] * ml * mask[b * SEQL + m];
        }
        alpha[tid] = s / len;
    }
    __syncthreads();
    if (tid == 0) {
        float mx = -1e30f;
        for (int l = 0; l < SEQL; ++l) {
            float a = (mask[b * SEQL + l] > 0.f) ? alpha[l] : -1e9f;
            alpha[l] = a;
            mx = fmaxf(mx, a);
        }
        float sum = 0.f;
        for (int l = 0; l < SEQL; ++l) { float e = __expf(alpha[l] - mx); beta[l] = e; sum += e; }
        inv_sum = 1.f / sum;
    }
    __syncthreads();
    if (tid < EMBD) {
        float inv = inv_sum, a = 0.f;
        const float* sb = seqf + (long long)b * LPAD * EMBD;
        for (int l = 0; l < SEQL; ++l) a += beta[l] * inv * sb[l * EMBD + tid];
        seqh[b * EMBD + tid] = a;
        acc2[b * EMBD + tid] = a;
    }
}

__global__ void k_convert_pad(const float* __restrict__ X, _Float16* __restrict__ Y,
                              int R, int Cc, int ldy) {
    long long i = (long long)blockIdx.x * blockDim.x + threadIdx.x;
    if (i >= (long long)R * Cc) return;
    int r = (int)(i / Cc), c = (int)(i % Cc);
    Y[(long long)r * ldy + c] = (_Float16)X[i];
}

__global__ void k_transpose_f16(const float* __restrict__ X, _Float16* __restrict__ Y,
                                int R, int Cc, int ldy) {
    long long i = (long long)blockIdx.x * blockDim.x + threadIdx.x;
    if (i >= (long long)R * Cc) return;
    int r = (int)(i / Cc), c = (int)(i % Cc);
    Y[(long long)c * ldy + r] = (_Float16)X[(long long)r * Cc + c];
}

// acc2 += s2 / max(||s2||, 1e-12); optionally emit out = acc2 * outscale
__global__ void k_norm_acc(const float* __restrict__ s2, float* __restrict__ acc2,
                           float* __restrict__ outp, float outscale) {
    int b = blockIdx.x, tid = threadIdx.x;
    __shared__ float part[128];
    float v = (tid < EMBD) ? s2[b * EMBD + tid] : 0.f;
    part[tid] = v * v;
    __syncthreads();
    for (int off = 64; off > 0; off >>= 1) {
        if (tid < off) part[tid] += part[tid + off];
        __syncthreads();
    }
    float nrm = sqrtf(part[0]);
    if (nrm < 1e-12f) nrm = 1e-12f;
    if (tid < EMBD) {
        float a = acc2[b * EMBD + tid] + v / nrm;
        acc2[b * EMBD + tid] = a;
        if (outp) outp[b * EMBD + tid] = a * outscale;
    }
}

// ---------------- host orchestration ----------------
static inline int cdiv(long long a, long long b) { return (int)((a + b - 1) / b); }

extern "C" void kernel_launch(void* const* d_in, const int* in_sizes, int n_in,
                              void* d_out, int out_size, void* d_ws, size_t ws_size,
                              hipStream_t stream) {
    const float* emb0      = (const float*)d_in[0];
    const float* adj_vals  = (const float*)d_in[1];
    const float* W_q       = (const float*)d_in[2];
    const float* W_k       = (const float*)d_in[3];
    const float* w_sess    = (const float*)d_in[4];
    const float* Dm        = (const float*)d_in[5];
    const float* Am        = (const float*)d_in[6];
    const int*   adj_rows  = (const int*)d_in[7];
    const int*   adj_cols  = (const int*)d_in[8];
    const int*   sess_item = (const int*)d_in[9];
    const int*   sess_len  = (const int*)d_in[10];
    const float* mask      = (const float*)d_in[11];
    float* out = (float*)d_out;

    char* w = (char*)d_ws;
    auto alloc = [&](long long bytes) { char* p = w; w += bytes; return (void*)p; };

    // --- zeroed prefix (pads + atomic accumulators) ---
    float*     emb1  = (float*)    alloc((long long)N_NODE * EMBD * 4);
    float*     emb2  = (float*)    alloc((long long)N_NODE * EMBD * 4);
    _Float16*  seq16 = (_Float16*) alloc((long long)BATCH * LPAD * KPAD * 2);
    _Float16*  WqT16 = (_Float16*) alloc((long long)EMBD * KPAD * 2);
    _Float16*  WkT16 = (_Float16*) alloc((long long)EMBD * KPAD * 2);
    _Float16*  wT16  = (_Float16*) alloc(2LL * EMBD * KPAD * 2);
    _Float16*  Q16   = (_Float16*) alloc((long long)BATCH * LPAD * KPAD * 2);
    _Float16*  K16   = (_Float16*) alloc((long long)BATCH * LPAD * KPAD * 2);
    _Float16*  s16   = (_Float16*) alloc((long long)BATCH * KPAD * 2);
    long long zero_bytes = (char*)w - (char*)d_ws;
    // --- rest ---
    float*     item  = (float*)    alloc((long long)N_NODE * EMBD * 4);
    float*     seqf  = (float*)    alloc((long long)BATCH * LPAD * EMBD * 4);
    float*     Cbuf  = (float*)    alloc((long long)BATCH * LPAD * LPAD * 4);
    float*     seqh  = (float*)    alloc((long long)BATCH * EMBD * 4);
    float*     acc2  = (float*)    alloc((long long)BATCH * EMBD * 4);
    _Float16*  D16   = (_Float16*) alloc((long long)BATCH * BATCH * 2);
    _Float16*  AT16  = (_Float16*) alloc((long long)BATCH * BATCH * 2);
    float*     DA    = (float*)    alloc((long long)BATCH * BATCH * 4);
    _Float16*  DA16  = (_Float16*) alloc((long long)BATCH * BATCH * 2);
    float*     s1    = (float*)    alloc((long long)BATCH * EMBD * 4);
    _Float16*  s1T16 = (_Float16*) alloc((long long)EMBD * BATCH * 2);
    float*     s2    = (float*)    alloc((long long)BATCH * EMBD * 4);

    // 1) zero pads/accumulators
    long long n16 = zero_bytes / 16;
    k_zero16<<<cdiv(n16, 256), 256, 0, stream>>>((float4*)d_ws, n16);

    // 2) HyperConv: two scatter-add SpMM layers + mean
    long long spwork = (long long)NNZE * 28;
    k_spmm<<<cdiv(spwork, 256), 256, 0, stream>>>(emb0, adj_vals, adj_rows, adj_cols, emb1);
    k_spmm<<<cdiv(spwork, 256), 256, 0, stream>>>(emb1, adj_vals, adj_rows, adj_cols, emb2);
    k_combine<<<cdiv((long long)N_NODE * EMBD, 256), 256, 0, stream>>>(emb0, emb1, emb2, item);

    // 3) stage f16 operands
    k_pack_weights<<<cdiv(EMBD * EMBD, 256), 256, 0, stream>>>(W_q, W_k, w_sess, WqT16, WkT16, wT16);
    k_convert_pad<<<cdiv((long long)BATCH * BATCH, 256), 256, 0, stream>>>(Dm, D16, BATCH, BATCH, BATCH);
    k_transpose_f16<<<cdiv((long long)BATCH * BATCH, 256), 256, 0, stream>>>(Am, AT16, BATCH, BATCH, BATCH);
    k_gather_seq<<<BATCH * SEQL, 128, 0, stream>>>(item, sess_item, seqf, seq16);

    // 4) WMMA projections: Q/K = sigmoid(seq @ W) -> f16 [B][64][128]
    int Mproj = BATCH * LPAD;                       // 32768
    int wavesP = (Mproj / 16) * (EMBD / 16);        // 14336
    k_wmma_gemm<<<cdiv(wavesP, 8), 256, 0, stream>>>(seq16, KPAD, 0, WqT16, KPAD, 0,
        (float*)nullptr, 0, 0, Q16, KPAD, 0, Mproj, EMBD, KPAD, 1, 1, 1.f);
    k_wmma_gemm<<<cdiv(wavesP, 8), 256, 0, stream>>>(seq16, KPAD, 0, WkT16, KPAD, 0,
        (float*)nullptr, 0, 0, K16, KPAD, 0, Mproj, EMBD, KPAD, 1, 1, 1.f);

    // 5) batched WMMA attention: C = sigmoid(Q K^T) / sqrt(d)
    int wavesC = BATCH * (LPAD / 16) * (LPAD / 16); // 8192
    float inv_sqrt_d = 1.f / sqrtf((float)EMBD);
    k_wmma_gemm<<<cdiv(wavesC, 8), 256, 0, stream>>>(Q16, KPAD, (long long)LPAD * KPAD,
        K16, KPAD, (long long)LPAD * KPAD, Cbuf, LPAD, (long long)LPAD * LPAD,
        (_Float16*)nullptr, 0, 0, LPAD, LPAD, KPAD, BATCH, 1, inv_sqrt_d);

    // 6) mask + alpha + softmax + pooling
    k_attn_post<<<BATCH, 128, 0, stream>>>(Cbuf, seqf, mask, sess_len, seqh, acc2);

    // 7) WMMA DA = D @ A (f32 + f16 outputs)
    int wavesDA = (BATCH / 16) * (BATCH / 16);      // 1024
    k_wmma_gemm<<<cdiv(wavesDA, 8), 256, 0, stream>>>(D16, BATCH, 0, AT16, BATCH, 0,
        DA, BATCH, 0, DA16, BATCH, 0, BATCH, BATCH, BATCH, 1, 0, 1.f);

    // 8) SessConv iterations
    int wavesS = (BATCH / 16) * (EMBD / 16);        // 224
    const float* s_src = seqh;
    for (int i = 0; i < 2; ++i) {
        k_convert_pad<<<cdiv((long long)BATCH * EMBD, 256), 256, 0, stream>>>(s_src, s16, BATCH, EMBD, KPAD);
        k_wmma_gemm<<<cdiv(wavesS, 8), 256, 0, stream>>>(s16, KPAD, 0, wT16 + (long long)i * EMBD * KPAD, KPAD, 0,
            s1, EMBD, 0, (_Float16*)nullptr, 0, 0, BATCH, EMBD, KPAD, 1, 0, 1.f);
        k_transpose_f16<<<cdiv((long long)BATCH * EMBD, 256), 256, 0, stream>>>(s1, s1T16, BATCH, EMBD, BATCH);
        k_wmma_gemm<<<cdiv(wavesS, 8), 256, 0, stream>>>(DA16, BATCH, 0, s1T16, BATCH, 0,
            s2, EMBD, 0, (_Float16*)nullptr, 0, 0, BATCH, EMBD, BATCH, 1, 0, 1.f);
        k_norm_acc<<<BATCH, 128, 0, stream>>>(s2, acc2, (i == 1) ? out : (float*)nullptr, 1.f / 3.f);
        s_src = s2;
    }
    (void)in_sizes; (void)n_in; (void)out_size; (void)ws_size;
}
